// mGRUCell_53987738911238
// MI455X (gfx1250) — matrix operand order repaired and checked by reading, not compiled, not run on hardware
//
#include <hip/hip_runtime.h>

typedef __attribute__((ext_vector_type(16))) __bf16 v16bf;
typedef __attribute__((ext_vector_type(8)))  float  v8f;

#define I_DIM   96
#define N_DIMV  64
#define BATCH_N 4096
#define HSTR    (I_DIM * N_DIMV)   // 6144 floats per batch row of h
#define HT_PITCH 68                // f32 pitch for 16x64 LDS tiles (16B-aligned rows, no bank conflicts)
#define U_PITCH  66                // bf16 pitch for transposed U in LDS (odd dword stride -> conflict-free)

union BFrag { unsigned u[8]; v16bf v; };

__device__ __forceinline__ float sigmoid_f(float x) {
    // exp overflow-safe: x<<0 -> exp(-x)=inf -> 0 ; x>>0 -> exp(-x)=0 -> 1
    return 1.0f / (1.0f + __expf(-x));
}
__device__ __forceinline__ float tanh_f(float x) {
    // 1 - 2/(e^{2x}+1): e=inf -> 1, e=0 -> -1 (no inf/inf NaN)
    float e = __expf(2.0f * x);
    return 1.0f - 2.0f / (e + 1.0f);
}

// B-matrix fragment (32x16 bf16, K-chunk c, N-tile t) from transposed U in LDS:
// element e of lane L holds U[k = 32c + 8*(L/16) + (e<8 ? e : e+8)][n = 16t + L%16]
__device__ __forceinline__ v16bf load_bfrag(const __bf16* sU, int t, int l16, int hlf, int c) {
    int n  = 16 * t + l16;
    int kb = 32 * c + 8 * hlf;
    const unsigned* p = (const unsigned*)(sU + n * U_PITCH + kb); // 4B aligned (kb even)
    BFrag f;
#pragma unroll
    for (int e = 0; e < 8; ++e) f.u[e] = p[e < 4 ? e : e + 4];
    return f.v;
}

__global__ __launch_bounds__(256, 1)
void mgru_cell_kernel(const float* __restrict__ X,  const float* __restrict__ h,
                      const float* __restrict__ Wz, const float* __restrict__ Wh,
                      const float* __restrict__ Uz, const float* __restrict__ Uh,
                      const float* __restrict__ bz, const float* __restrict__ bh,
                      float* __restrict__ out)
{
    __shared__ __align__(16) __bf16 sUz[N_DIMV * U_PITCH];        // U_z[i] transposed: [n][k]
    __shared__ __align__(16) __bf16 sUh[N_DIMV * U_PITCH];        // U_h[i] transposed
    __shared__ __align__(16) float  sH [8][16 * HT_PITCH];        // per-wave 16x64 h tile (f32)
    __shared__ __align__(16) float  sA2[8][16 * HT_PITCH];        // per-wave z*h3 staging

    const int tid  = threadIdx.x;
    const int lane = tid & 31;
    const int wv   = tid >> 5;
    const int i    = blockIdx.y;
    const int b0   = blockIdx.x * 128 + wv * 16;
    const int hlf  = lane >> 4;     // lane half (0/1)
    const int l16  = lane & 15;

    // ---- stage U_z[i], U_h[i] into LDS, transposed, bf16 (coalesced global reads over n)
    const float* Uzi = Uz + (size_t)i * N_DIMV * N_DIMV;
    const float* Uhi = Uh + (size_t)i * N_DIMV * N_DIMV;
#pragma unroll
    for (int e = 0; e < 16; ++e) {
        int idx = e * 256 + tid;                // 0..4095
        int k = idx >> 6, n = idx & 63;
        sUz[n * U_PITCH + k] = (__bf16)Uzi[idx];
        sUh[n * U_PITCH + k] = (__bf16)Uhi[idx];
    }

    // ---- stage this wave's 16x64 tile of h (f32, float4 global loads)
    const float* hbase = h + (size_t)b0 * HSTR + (size_t)i * N_DIMV;
#pragma unroll
    for (int v = 0; v < 8; ++v) {
        int off = v * 128 + lane * 4;           // [0,1024)
        int r = off >> 6, c = off & 63;
        float4 d = *(const float4*)(hbase + (size_t)r * HSTR + c);
        *(float4*)&sH[wv][r * HT_PITCH + c] = d;
    }
    __syncthreads();

    // ---- A1 fragments (16x32 bf16 each): lane L row m=L%16,
    //      element e -> k = 32c + 8*(L/16) + (e<8 ? e : e+8)
    v16bf a1[2];
    {
        const float* hrow = &sH[wv][l16 * HT_PITCH];
#pragma unroll
        for (int c = 0; c < 2; ++c) {
            int kb = 32 * c + 8 * hlf;
#pragma unroll
            for (int e = 0; e < 8; ++e) {
                a1[c][e]     = (__bf16)hrow[kb + e];
                a1[c][e + 8] = (__bf16)hrow[kb + 16 + e];
            }
        }
    }

    // ---- GEMM1: acc1 = h3 @ U_z[i]   (8x v_wmma_f32_16x16x32_bf16)
    v8f acc1[4];
#pragma unroll
    for (int t = 0; t < 4; ++t) {
        v8f c = {};
#pragma unroll
        for (int kc = 0; kc < 2; ++kc) {
            v16bf b = load_bfrag(sUz, t, l16, hlf, kc);
            c = __builtin_amdgcn_wmma_f32_16x16x32_bf16(false, a1[kc], false, b,
                                                        (short)0, c, false, false);
        }
        acc1[t] = c;
    }

    // ---- per-lane gate scalars
    float xv[8];
#pragma unroll
    for (int g = 0; g < 8; ++g)
        xv[g] = X[(size_t)(b0 + g + 8 * hlf) * I_DIM + i];
    float wzv[4], bzv[4], whv[4], bhv[4];
#pragma unroll
    for (int t = 0; t < 4; ++t) {
        int n = 16 * t + l16;
        wzv[t] = Wz[i * N_DIMV + n];  bzv[t] = bz[i * N_DIMV + n];
        whv[t] = Wh[i * N_DIMV + n];  bhv[t] = bh[i * N_DIMV + n];
    }

    // ---- z = sigmoid(.); stage z*h3 for GEMM2 (C/D layout: lane holds (m = g+8*hlf, n))
    float zval[4][8], h3v[4][8];
#pragma unroll
    for (int t = 0; t < 4; ++t) {
        int n = 16 * t + l16;
#pragma unroll
        for (int g = 0; g < 8; ++g) {
            int m = g + 8 * hlf;
            float z  = sigmoid_f(acc1[t][g] + xv[g] * wzv[t] + bzv[t]);
            float h3 = sH[wv][m * HT_PITCH + n];
            zval[t][g] = z;
            h3v[t][g]  = h3;
            sA2[wv][m * HT_PITCH + n] = z * h3;
        }
    }
    __syncthreads();

    // ---- A2 fragments from staged z*h3
    v16bf a2[2];
    {
        const float* arow = &sA2[wv][l16 * HT_PITCH];
#pragma unroll
        for (int c = 0; c < 2; ++c) {
            int kb = 32 * c + 8 * hlf;
#pragma unroll
            for (int e = 0; e < 8; ++e) {
                a2[c][e]     = (__bf16)arow[kb + e];
                a2[c][e + 8] = (__bf16)arow[kb + 16 + e];
            }
        }
    }

    float* outN = out + (size_t)b0 * HSTR + (size_t)i * N_DIMV;           // h_new
    float* outT = outN + (size_t)BATCH_N * HSTR;                          // h_tilde

    // ---- GEMM2 + gates + stores
#pragma unroll
    for (int t = 0; t < 4; ++t) {
        v8f c = {};
#pragma unroll
        for (int kc = 0; kc < 2; ++kc) {
            v16bf b = load_bfrag(sUh, t, l16, hlf, kc);
            c = __builtin_amdgcn_wmma_f32_16x16x32_bf16(false, a2[kc], false, b,
                                                        (short)0, c, false, false);
        }
        int n = 16 * t + l16;
#pragma unroll
        for (int g = 0; g < 8; ++g) {
            int m  = g + 8 * hlf;
            float ht = tanh_f(c[g] + xv[g] * whv[t] + bhv[t]);
            float z  = zval[t][g];
            float hn = z * h3v[t][g] + (1.0f - z) * ht;
            outN[(size_t)m * HSTR + n] = hn;
            outT[(size_t)m * HSTR + n] = ht;
        }
    }
}

extern "C" void kernel_launch(void* const* d_in, const int* in_sizes, int n_in,
                              void* d_out, int out_size, void* d_ws, size_t ws_size,
                              hipStream_t stream) {
    const float* X  = (const float*)d_in[0];
    const float* h  = (const float*)d_in[1];
    const float* Wz = (const float*)d_in[2];
    const float* Wh = (const float*)d_in[3];
    const float* Uz = (const float*)d_in[4];
    const float* Uh = (const float*)d_in[5];
    const float* bz = (const float*)d_in[6];
    const float* bh = (const float*)d_in[7];
    float* out = (float*)d_out;

    dim3 grid(BATCH_N / 128, I_DIM, 1);   // 32 x 96 blocks
    mgru_cell_kernel<<<grid, 256, 0, stream>>>(X, h, Wz, Wh, Uz, Uh, bz, bh, out);
}
